// LinearSelfAttention_58944131170446
// MI455X (gfx1250) — compile-verified, matrix operand order checked by reading, not run on hardware
//
#include <hip/hip_runtime.h>
#include <hip/hip_bf16.h>
#include <stdint.h>

// Problem constants (B=4, T=4096, D=1024, H=16, hd=64)
#define B_   4
#define T_   4096
#define D_   1024
#define H_   16
#define HD_  64
#define M_   (B_ * T_)     // 16384 rows
#define N3_  (3 * D_)      // 3072
#define SCALE_ 0.125f      // 64^-0.5
#define EPS_   1e-8f

#if __has_builtin(__builtin_amdgcn_tensor_load_to_lds) && \
    __has_builtin(__builtin_amdgcn_s_wait_tensorcnt)
#define HAVE_TDM 1
#else
#define HAVE_TDM 0
#endif

typedef __attribute__((ext_vector_type(16))) __bf16 bf16x16;
typedef __attribute__((ext_vector_type(8)))  __bf16 bf16x8;
typedef __attribute__((ext_vector_type(8)))  float  f32x8;

__device__ __forceinline__ unsigned short f2bf(float f) {
    union { float f; uint32_t u; } v; v.f = f;
    uint32_t u = v.u;
    uint32_t r = (u + 0x7FFFu + ((u >> 16) & 1u)) >> 16;  // round-to-nearest-even
    return (unsigned short)r;
}
__device__ __forceinline__ float bf2f(unsigned short h) {
    union { uint32_t u; float f; } v; v.u = ((uint32_t)h) << 16;
    return v.f;
}
__device__ __forceinline__ f32x8 zero8() {
    f32x8 z;
#pragma unroll
    for (int i = 0; i < 8; ++i) z[i] = 0.f;
    return z;
}

// A/B fragment for V_WMMA_F32_16X16X32_BF16 (CDNA5 ISA, 16-bit A 16x32):
// lane L: row = L&15, K = (L>>4)*8 + j (VGPRs 0-3) and +16 (VGPRs 4-7).
__device__ __forceinline__ bf16x16 load_frag(const unsigned short* base, int ld, int lane) {
    int r  = lane & 15;
    int kh = (lane >> 4) << 3;
    const unsigned short* p = base + (size_t)r * ld + kh;
    bf16x8 lo = *reinterpret_cast<const bf16x8*>(p);
    bf16x8 hi = *reinterpret_cast<const bf16x8*>(p + 16);
    return __builtin_shufflevector(lo, hi, 0,1,2,3,4,5,6,7,8,9,10,11,12,13,14,15);
}

__device__ __forceinline__ f32x8 wmma_bf16(bf16x16 a, bf16x16 b, f32x8 c) {
    return __builtin_amdgcn_wmma_f32_16x16x32_bf16(false, a, false, b, (short)0, c, false, false);
}

#define LDT 40  // padded LDS row stride (elements): 32 data + 8 pad

#if HAVE_TDM
typedef __attribute__((ext_vector_type(4))) unsigned int u32x4_t;
typedef __attribute__((ext_vector_type(8))) int i32x8_t;
typedef __attribute__((ext_vector_type(4))) int i32x4_t;

// TDM: load a 128-row x 32-elem bf16 tile (row pitch ldElems) from global to
// LDS with 4-DWORD padding after every 16 DWORDs -> padded LDS stride LDT=40.
// D# layout per CDNA5 ISA. Tracked by TENSORcnt.
// This toolchain exposes the 6-arg builtin: (g0, g1, g2, g3, g4, cpol).
__device__ __forceinline__ void tdm_load_tile(const unsigned short* gsrc,
                                              unsigned int ldElems,
                                              unsigned short* ldsDst) {
    unsigned long long ga = (unsigned long long)(uintptr_t)gsrc;
    u32x4_t g0;
    g0[0] = 1u;                                   // count=1, user descriptor
    g0[1] = (unsigned int)(uintptr_t)ldsDst;      // lds_addr (byte offset)
    g0[2] = (unsigned int)ga;                     // global_addr[31:0]
    g0[3] = (unsigned int)((ga >> 32) & 0x01FFFFFFu) | (2u << 30);  // [56:32] | type=2
    i32x8_t g1;
    g1[0] = (int)((1u << 16)       // data_size = 2 bytes
                | (1u << 20)       // pad_enable
                | (3u << 22)       // pad_interval: 16 DWORDs (one 64B row)
                | (3u << 25));     // pad_amount: 4 DWORDs (8 elements)
    g1[1] = (int)((ldElems & 0xFFFFu) << 16);                      // tensor_dim0[15:0]
    g1[2] = (int)((ldElems >> 16) | (128u << 16));                 // dim0[31:16] | tensor_dim1[15:0]=128
    g1[3] = (int)(32u << 16);                                      // dim1[31:16]=0 | tile_dim0=32
    g1[4] = 128;                                                   // tile_dim1=128, tile_dim2=0
    g1[5] = (int)ldElems;                                          // tensor_dim0_stride[31:0]
    g1[6] = 0;
    g1[7] = 0;
    i32x4_t z4 = {0, 0, 0, 0};
    i32x8_t z8 = {0, 0, 0, 0, 0, 0, 0, 0};
    __builtin_amdgcn_tensor_load_to_lds(g0, g1, z4, z4, z8, 0);
}
#endif

// ---------------------------------------------------------------- cast fp32->bf16
__global__ void k_cast(const float* __restrict__ src, unsigned short* __restrict__ dst, int n) {
    int i = blockIdx.x * blockDim.x + threadIdx.x;
    if (i < n) dst[i] = f2bf(src[i]);
}

// ---------------------------------------------------------------- RMSNorm + cast
__global__ __launch_bounds__(256) void k_rmsnorm(const float* __restrict__ x,
                                                 const float* __restrict__ w,
                                                 unsigned short* __restrict__ xn) {
    int row = blockIdx.x;
    int tid = threadIdx.x;
    const float* xr = x + (size_t)row * D_;
    float v[4]; float s = 0.f;
#pragma unroll
    for (int i = 0; i < 4; ++i) { v[i] = xr[tid + 256 * i]; s += v[i] * v[i]; }
    __shared__ float red[256];
    red[tid] = s; __syncthreads();
    for (int off = 128; off > 0; off >>= 1) {
        if (tid < off) red[tid] += red[tid + off];
        __syncthreads();
    }
    float norm = sqrtf(red[0]) * 0.03125f;        // ||x|| * D^-0.5
    float inv  = 1.f / (norm + EPS_);
    unsigned short* xo = xn + (size_t)row * D_;
#pragma unroll
    for (int i = 0; i < 4; ++i) {
        int c = tid + 256 * i;
        xo[c] = f2bf(w[c] * v[i] * inv);
    }
}

// ---------------------------------------------------------------- QKV GEMM
// C[16384,3072] = xn @ w_qkv^T. 128x128 tile; TDM double-buffered LDS staging;
// 8 waves x (2x4) WMMA frags. Epilogue scatters q->[bh,t,d], k,v->[bh,d,t].
__global__ __launch_bounds__(256) void k_qkv(const unsigned short* __restrict__ A,
                                             const unsigned short* __restrict__ W,
                                             unsigned short* __restrict__ qb,
                                             unsigned short* __restrict__ kb,
                                             unsigned short* __restrict__ vb) {
    __shared__ __align__(128) unsigned short As[2 * 128 * LDT];
    __shared__ __align__(128) unsigned short Bs[2 * 128 * LDT];
    int tid  = threadIdx.x;
    int lane = tid & 31, wid = tid >> 5;
    int mBlk = blockIdx.y * 128, nBlk = blockIdx.x * 128;
    int mW = (wid & 3) * 32, nW = (wid >> 2) * 64;

    f32x8 acc[2][4];
#pragma unroll
    for (int i = 0; i < 2; ++i)
#pragma unroll
        for (int j = 0; j < 4; ++j) acc[i][j] = zero8();

    const int NS = D_ / 32;
#if HAVE_TDM
    if (wid == 0) {
        tdm_load_tile(A + (size_t)mBlk * D_, D_, As);
        tdm_load_tile(W + (size_t)nBlk * D_, D_, Bs);
    }
    for (int s = 0; s < NS; ++s) {
        if (wid == 0) {
            if (s + 1 < NS) {
                tdm_load_tile(A + (size_t)mBlk * D_ + (s + 1) * 32, D_,
                              As + ((s + 1) & 1) * (128 * LDT));
                tdm_load_tile(W + (size_t)nBlk * D_ + (s + 1) * 32, D_,
                              Bs + ((s + 1) & 1) * (128 * LDT));
                __builtin_amdgcn_s_wait_tensorcnt(2);   // in-order: tile s done
            } else {
                __builtin_amdgcn_s_wait_tensorcnt(0);
            }
        }
        __syncthreads();
        const unsigned short* Ab = As + (s & 1) * (128 * LDT);
        const unsigned short* Bb = Bs + (s & 1) * (128 * LDT);
        bf16x16 af[2], bfm[4];
#pragma unroll
        for (int i = 0; i < 2; ++i) af[i]  = load_frag(Ab + (mW + 16 * i) * LDT, LDT, lane);
#pragma unroll
        for (int j = 0; j < 4; ++j) bfm[j] = load_frag(Bb + (nW + 16 * j) * LDT, LDT, lane);
#pragma unroll
        for (int i = 0; i < 2; ++i)
#pragma unroll
            for (int j = 0; j < 4; ++j)
                acc[i][j] = wmma_bf16(af[i], bfm[j], acc[i][j]);
        __syncthreads();
    }
#else
    int srow = tid >> 1, scol = (tid & 1) * 16;
    for (int s = 0; s < NS; ++s) {
        int k0 = s * 32;
        const uint4* ga = reinterpret_cast<const uint4*>(A + (size_t)(mBlk + srow) * D_ + k0 + scol);
        const uint4* gb = reinterpret_cast<const uint4*>(W + (size_t)(nBlk + srow) * D_ + k0 + scol);
        uint4 a0 = ga[0], a1 = ga[1];
        uint4 b0 = gb[0], b1 = gb[1];
        uint4* da = reinterpret_cast<uint4*>(&As[srow * LDT + scol]);
        uint4* db = reinterpret_cast<uint4*>(&Bs[srow * LDT + scol]);
        da[0] = a0; da[1] = a1;
        db[0] = b0; db[1] = b1;
        __syncthreads();
        bf16x16 af[2], bfm[4];
#pragma unroll
        for (int i = 0; i < 2; ++i) af[i]  = load_frag(&As[(mW + 16 * i) * LDT], LDT, lane);
#pragma unroll
        for (int j = 0; j < 4; ++j) bfm[j] = load_frag(&Bs[(nW + 16 * j) * LDT], LDT, lane);
#pragma unroll
        for (int i = 0; i < 2; ++i)
#pragma unroll
            for (int j = 0; j < 4; ++j)
                acc[i][j] = wmma_bf16(af[i], bfm[j], acc[i][j]);
        __syncthreads();
    }
#endif

    // Epilogue: the q/k/v selector depends only on the fragment base column
    // (16-aligned, never crosses a 1024 boundary) -> wave-uniform scalar branch.
    int nL = lane & 15, mH = (lane >> 4) * 8;
    int b = mBlk >> 12;                       // block's batch (128 rows < T)
#pragma unroll
    for (int i = 0; i < 2; ++i) {
        int tU = (mBlk & (T_ - 1)) + mW + 16 * i + mH;   // + r below
#pragma unroll
        for (int j = 0; j < 4; ++j) {
            int nf   = nBlk + nW + 16 * j;   // uniform fragment base column
            int part = nf >> 10;             // 0=q 1=k 2=v   (scalar branch)
            int dd   = nf & 1023;
            int h = dd >> 6, dB = dd & 63;
            int bh = b * H_ + h;
            if (part == 0) {
                unsigned short* dst = qb + ((size_t)bh * T_) * HD_ + (dB + nL);
#pragma unroll
                for (int r = 0; r < 8; ++r)
                    dst[(size_t)(tU + r) * HD_] = f2bf(acc[i][j][r]);
            } else {
                unsigned short* base = (part == 1) ? kb : vb;
                unsigned short* dst = base + ((size_t)bh * HD_ + dB + nL) * T_ + tU;
#pragma unroll
                for (int r = 0; r < 8; ++r)
                    dst[r] = f2bf(acc[i][j][r]);
            }
        }
    }
}

// ---------------------------------------------------------------- softmax over T
__global__ __launch_bounds__(256) void k_softmax(unsigned short* __restrict__ kb) {
    int row = blockIdx.x;                       // 0..BH*64-1
    unsigned short* kr = kb + (size_t)row * T_;
    int tid = threadIdx.x;
    float v[16];
    float mx = -1e30f;
#pragma unroll
    for (int i = 0; i < 16; ++i) {
        v[i] = bf2f(kr[tid + 256 * i]) * SCALE_;
        mx = fmaxf(mx, v[i]);
    }
    __shared__ float red[256];
    red[tid] = mx; __syncthreads();
    for (int off = 128; off > 0; off >>= 1) {
        if (tid < off) red[tid] = fmaxf(red[tid], red[tid + off]);
        __syncthreads();
    }
    mx = red[0]; __syncthreads();
    float s = 0.f;
#pragma unroll
    for (int i = 0; i < 16; ++i) { v[i] = __expf(v[i] - mx); s += v[i]; }
    red[tid] = s; __syncthreads();
    for (int off = 128; off > 0; off >>= 1) {
        if (tid < off) red[tid] += red[tid + off];
        __syncthreads();
    }
    float inv = 1.f / red[0];
#pragma unroll
    for (int i = 0; i < 16; ++i) kr[tid + 256 * i] = f2bf(v[i] * inv);
}

// ---------------------------------------------------------------- context = knorm^T @ v
// Per head: [64 x T] @ [T x 64], K=4096 streaming. Output stored transposed
// ctx[bh, v, d] so it loads directly as a B-fragment in the next kernel.
__global__ __launch_bounds__(128) void k_ctx(const unsigned short* __restrict__ kn,
                                             const unsigned short* __restrict__ vv,
                                             unsigned short* __restrict__ ctx) {
    int bh = blockIdx.x;
    int tid = threadIdx.x, lane = tid & 31, wid = tid >> 5;   // 4 waves
    const unsigned short* Kb = kn + (size_t)bh * HD_ * T_;
    const unsigned short* Vb = vv + (size_t)bh * HD_ * T_;
    f32x8 acc[4];
#pragma unroll
    for (int j = 0; j < 4; ++j) acc[j] = zero8();

    for (int t0 = 0; t0 < T_; t0 += 32) {
        if (t0 + 128 < T_) {   // prefetch four tiles ahead -> global_prefetch_b8
            __builtin_prefetch(Kb + (size_t)(wid * 16) * T_ + t0 + 128, 0, 1);
            __builtin_prefetch(Vb + (size_t)(wid * 16) * T_ + t0 + 128, 0, 1);
        }
        bf16x16 a = load_frag(Kb + (size_t)(wid * 16) * T_ + t0, T_, lane);
#pragma unroll
        for (int j = 0; j < 4; ++j) {
            bf16x16 b = load_frag(Vb + (size_t)(j * 16) * T_ + t0, T_, lane);
            acc[j] = wmma_bf16(a, b, acc[j]);
        }
    }
    int nL = lane & 15, mH = (lane >> 4) * 8;
#pragma unroll
    for (int j = 0; j < 4; ++j)
#pragma unroll
        for (int r = 0; r < 8; ++r) {
            int m = wid * 16 + r + mH;      // d
            int n = j * 16 + nL;            // v
            ctx[((size_t)bh * HD_ + n) * HD_ + m] = f2bf(acc[j][r]);
        }
}

// ---------------------------------------------------------------- out = q @ context
__global__ __launch_bounds__(256) void k_qctx(const unsigned short* __restrict__ qb,
                                              const unsigned short* __restrict__ ctx,
                                              unsigned short* __restrict__ attn) {
    int bh = blockIdx.y; int b = bh >> 4, h = bh & 15;
    int mBlk = blockIdx.x * 256;
    int tid = threadIdx.x, lane = tid & 31, wid = tid >> 5;
    int mW = wid * 32;
    const unsigned short* Qb = qb + (size_t)bh * T_ * HD_;
    const unsigned short* Cb = ctx + (size_t)bh * HD_ * HD_;
    f32x8 acc[2][4];
#pragma unroll
    for (int i = 0; i < 2; ++i)
#pragma unroll
        for (int j = 0; j < 4; ++j) acc[i][j] = zero8();

#pragma unroll
    for (int k0 = 0; k0 < HD_; k0 += 32) {
        bf16x16 a0 = load_frag(Qb + (size_t)(mBlk + mW) * HD_ + k0, HD_, lane);
        bf16x16 a1 = load_frag(Qb + (size_t)(mBlk + mW + 16) * HD_ + k0, HD_, lane);
#pragma unroll
        for (int j = 0; j < 4; ++j) {
            bf16x16 bfr = load_frag(Cb + (size_t)(j * 16) * HD_ + k0, HD_, lane);
            acc[0][j] = wmma_bf16(a0, bfr, acc[0][j]);
            acc[1][j] = wmma_bf16(a1, bfr, acc[1][j]);
        }
    }
    int nL = lane & 15, mH = (lane >> 4) * 8;
#pragma unroll
    for (int i = 0; i < 2; ++i)
#pragma unroll
        for (int j = 0; j < 4; ++j)
#pragma unroll
            for (int r = 0; r < 8; ++r) {
                int t = mBlk + mW + 16 * i + r + mH;
                int n = 16 * j + nL;
                attn[((size_t)(b * T_ + t)) * D_ + h * HD_ + n] = f2bf(acc[i][j][r]);
            }
}

// ---------------------------------------------------------------- final proj + residual
__global__ __launch_bounds__(256) void k_final(const unsigned short* __restrict__ A,
                                               const unsigned short* __restrict__ W,
                                               const float* __restrict__ x,
                                               float* __restrict__ out) {
    __shared__ __align__(128) unsigned short As[2 * 128 * LDT];
    __shared__ __align__(128) unsigned short Bs[2 * 128 * LDT];
    int tid  = threadIdx.x;
    int lane = tid & 31, wid = tid >> 5;
    int mBlk = blockIdx.y * 128, nBlk = blockIdx.x * 128;
    int mW = (wid & 3) * 32, nW = (wid >> 2) * 64;

    f32x8 acc[2][4];
#pragma unroll
    for (int i = 0; i < 2; ++i)
#pragma unroll
        for (int j = 0; j < 4; ++j) acc[i][j] = zero8();

    const int NS = D_ / 32;
#if HAVE_TDM
    if (wid == 0) {
        tdm_load_tile(A + (size_t)mBlk * D_, D_, As);
        tdm_load_tile(W + (size_t)nBlk * D_, D_, Bs);
    }
    for (int s = 0; s < NS; ++s) {
        if (wid == 0) {
            if (s + 1 < NS) {
                tdm_load_tile(A + (size_t)mBlk * D_ + (s + 1) * 32, D_,
                              As + ((s + 1) & 1) * (128 * LDT));
                tdm_load_tile(W + (size_t)nBlk * D_ + (s + 1) * 32, D_,
                              Bs + ((s + 1) & 1) * (128 * LDT));
                __builtin_amdgcn_s_wait_tensorcnt(2);
            } else {
                __builtin_amdgcn_s_wait_tensorcnt(0);
            }
        }
        __syncthreads();
        const unsigned short* Ab = As + (s & 1) * (128 * LDT);
        const unsigned short* Bb = Bs + (s & 1) * (128 * LDT);
        bf16x16 af[2], bfm[4];
#pragma unroll
        for (int i = 0; i < 2; ++i) af[i]  = load_frag(Ab + (mW + 16 * i) * LDT, LDT, lane);
#pragma unroll
        for (int j = 0; j < 4; ++j) bfm[j] = load_frag(Bb + (nW + 16 * j) * LDT, LDT, lane);
#pragma unroll
        for (int i = 0; i < 2; ++i)
#pragma unroll
            for (int j = 0; j < 4; ++j)
                acc[i][j] = wmma_bf16(af[i], bfm[j], acc[i][j]);
        __syncthreads();
    }
#else
    int srow = tid >> 1, scol = (tid & 1) * 16;
    for (int s = 0; s < NS; ++s) {
        int k0 = s * 32;
        const uint4* ga = reinterpret_cast<const uint4*>(A + (size_t)(mBlk + srow) * D_ + k0 + scol);
        const uint4* gb = reinterpret_cast<const uint4*>(W + (size_t)(nBlk + srow) * D_ + k0 + scol);
        uint4 a0 = ga[0], a1 = ga[1];
        uint4 b0 = gb[0], b1 = gb[1];
        uint4* da = reinterpret_cast<uint4*>(&As[srow * LDT + scol]);
        uint4* db = reinterpret_cast<uint4*>(&Bs[srow * LDT + scol]);
        da[0] = a0; da[1] = a1;
        db[0] = b0; db[1] = b1;
        __syncthreads();
        bf16x16 af[2], bfm[4];
#pragma unroll
        for (int i = 0; i < 2; ++i) af[i]  = load_frag(&As[(mW + 16 * i) * LDT], LDT, lane);
#pragma unroll
        for (int j = 0; j < 4; ++j) bfm[j] = load_frag(&Bs[(nW + 16 * j) * LDT], LDT, lane);
#pragma unroll
        for (int i = 0; i < 2; ++i)
#pragma unroll
            for (int j = 0; j < 4; ++j)
                acc[i][j] = wmma_bf16(af[i], bfm[j], acc[i][j]);
        __syncthreads();
    }
#endif

    int nL = lane & 15, mH = (lane >> 4) * 8;
#pragma unroll
    for (int i = 0; i < 2; ++i)
#pragma unroll
        for (int j = 0; j < 4; ++j)
#pragma unroll
            for (int r = 0; r < 8; ++r) {
                int m = mBlk + mW + 16 * i + r + mH;
                int n = nBlk + nW + 16 * j + nL;
                out[(size_t)m * D_ + n] = acc[i][j][r] + x[(size_t)m * D_ + n];
            }
}

// ----------------------------------------------------------------
extern "C" void kernel_launch(void* const* d_in, const int* in_sizes, int n_in,
                              void* d_out, int out_size, void* d_ws, size_t ws_size,
                              hipStream_t stream) {
    (void)in_sizes; (void)n_in; (void)out_size; (void)ws_size;
    const float* x    = (const float*)d_in[0];
    const float* rw   = (const float*)d_in[1];
    const float* wqkv = (const float*)d_in[2];
    const float* wout = (const float*)d_in[3];
    float* out = (float*)d_out;

    char* p = (char*)d_ws;
    auto take = [&](size_t bytes) {
        char* r = p; p += (bytes + 255) & ~(size_t)255; return r;
    };
    unsigned short* xn   = (unsigned short*)take((size_t)M_ * D_ * 2);   // 32 MB
    unsigned short* wq_b = (unsigned short*)take((size_t)N3_ * D_ * 2);  //  6 MB
    unsigned short* wo_b = (unsigned short*)take((size_t)D_ * D_ * 2);   //  2 MB
    unsigned short* qb   = (unsigned short*)take((size_t)M_ * D_ * 2);   // 32 MB [bh,t,d]
    unsigned short* kb   = (unsigned short*)take((size_t)M_ * D_ * 2);   // 32 MB [bh,d,t]
    unsigned short* vb   = (unsigned short*)take((size_t)M_ * D_ * 2);   // 32 MB [bh,d,t]
    unsigned short* ctx  = (unsigned short*)take((size_t)B_ * H_ * HD_ * HD_ * 2);
    unsigned short* attn = (unsigned short*)take((size_t)M_ * D_ * 2);   // 32 MB [b,t,D]

    k_cast<<<(N3_ * D_ + 255) / 256, 256, 0, stream>>>(wqkv, wq_b, N3_ * D_);
    k_cast<<<(D_ * D_ + 255) / 256, 256, 0, stream>>>(wout, wo_b, D_ * D_);
    k_rmsnorm<<<M_, 256, 0, stream>>>(x, rw, xn);

    dim3 g1(N3_ / 128, M_ / 128);
    k_qkv<<<g1, 256, 0, stream>>>(xn, wq_b, qb, kb, vb);

    k_softmax<<<B_ * H_ * HD_, 256, 0, stream>>>(kb);

    k_ctx<<<B_ * H_, 128, 0, stream>>>(kb, vb, ctx);

    dim3 g2(T_ / 256, B_ * H_);
    k_qctx<<<g2, 256, 0, stream>>>(qb, ctx, attn);

    dim3 g3(D_ / 128, M_ / 128);
    k_final<<<g3, 256, 0, stream>>>(attn, wo_b, x, out);
}
